// WindowAttention_33646773796919
// MI455X (gfx1250) — matrix operand order repaired and checked by reading, not compiled, not run on hardware
//
#include <hip/hip_runtime.h>

typedef __bf16 bf16;
typedef __bf16 v16bf __attribute__((ext_vector_type(16)));
typedef __bf16 v8bf  __attribute__((ext_vector_type(8)));
typedef __bf16 v4bf  __attribute__((ext_vector_type(4)));
typedef __bf16 v2bf  __attribute__((ext_vector_type(2)));
typedef float  v8f   __attribute__((ext_vector_type(8)));
typedef float  v4f   __attribute__((ext_vector_type(4)));

#define N_TOK  49
#define NPAD   64
#define CDIM   128
#define HEADS  4
#define HDIM   32
#define NW     64
#define BWIN   4096
#define NN     (N_TOK * N_TOK)          // 2401
#define QK_SCALE 0.17677669529663687f   // 32^-0.5

// workspace layout (bytes):
//   [0,      98304)   wqkvT  bf16 [384][128]         (transposed qkv_w)
//   [98304, 131072)   wprojT bf16 [128][128]         (transposed proj_w)
//   [131072,2589696)  cmb    f32  [64][4][49*49]     (rel-pos bias + window mask, combined)

__global__ void prep_kernel(const float* __restrict__ qkv_w,
                            const float* __restrict__ proj_w,
                            const float* __restrict__ bias_table,
                            const int*   __restrict__ rel_index,
                            const float* __restrict__ mask,
                            bf16* __restrict__ wqkvT,
                            bf16* __restrict__ wprojT,
                            float* __restrict__ cmb) {
  int i = blockIdx.x * blockDim.x + threadIdx.x;
  if (i < 384 * 128) {
    int n = i >> 7, k = i & 127;
    wqkvT[i] = (bf16)qkv_w[k * 384 + n];
  } else if (i < 384 * 128 + 128 * 128) {
    int j = i - 384 * 128;
    int n = j >> 7, k = j & 127;
    wprojT[j] = (bf16)proj_w[k * 128 + n];
  } else if (i < 384 * 128 + 128 * 128 + NW * HEADS * NN) {
    int j = i - (384 * 128 + 128 * 128);
    int w   = j / (HEADS * NN);
    int rem = j % (HEADS * NN);
    int h   = rem / NN;
    int p   = rem % NN;
    cmb[j] = bias_table[rel_index[p] * HEADS + h] + mask[w * NN + p];
  }
}

__device__ __forceinline__ v8f v8f_zero() {
  v8f z;
#pragma unroll
  for (int i = 0; i < 8; ++i) z[i] = 0.f;
  return z;
}

__device__ __forceinline__ v8f wmma_bf16(v16bf a, v16bf b, v8f c) {
  return __builtin_amdgcn_wmma_f32_16x16x32_bf16(false, a, false, b,
                                                 (short)0, c, false, false);
}

// A fragment: 16 rows x 32 K. row = lane&15.
// lanes 0-15: K = 0..15; lanes 16-31: K = 16..31 (contiguous in k).
// element (row,k) at base[row*stride + k].
__device__ __forceinline__ v16bf load_a(const bf16* base, int stride,
                                        int row0, int k0, int lo, int hi) {
  const bf16* p = base + (row0 + lo) * stride + k0 + hi * 16;
  v8bf l = *(const v8bf*)p;
  v8bf h = *(const v8bf*)(p + 8);
  return __builtin_shufflevector(l, h, 0, 1, 2, 3, 4, 5, 6, 7,
                                 8, 9, 10, 11, 12, 13, 14, 15);
}

// B fragment: 32 K x 16 cols. col = lane&15.
// lanes 0-15 hold K=0..15, lanes 16-31 hold K=16..31 (contiguous in k).
// element (k,n) at base[n*stride + k].
__device__ __forceinline__ v16bf load_b(const bf16* base, int stride,
                                        int n0, int k0, int lo, int hi) {
  const bf16* p = base + (n0 + lo) * stride + k0 + hi * 16;
  return *(const v16bf*)p;
}

__launch_bounds__(128)
__global__ void win_attn_kernel(const float* __restrict__ x,
                                const float* __restrict__ qkv_b,
                                const float* __restrict__ proj_b,
                                const bf16*  __restrict__ wqkvT,
                                const bf16*  __restrict__ wprojT,
                                const float* __restrict__ cmb,
                                float* __restrict__ out) {
  // 64 KB total: 5 workgroups per 320 KB WGP
  __shared__ __align__(32) bf16 sX[NPAD * CDIM];         // 16 KB; reused as P scratch
  __shared__ __align__(32) bf16 sQ[HEADS][NPAD * HDIM];  // 16 KB; reused as O (head-major)
  __shared__ __align__(32) bf16 sK[HEADS][NPAD * HDIM];  // 16 KB
  __shared__ __align__(32) bf16 sVT[HEADS][HDIM * NPAD]; // 16 KB (V transposed)

  const int tid  = threadIdx.x;
  const int lane = tid & 31;
  const int wv   = tid >> 5;    // wave id 0..3; also this wave's row-tile & head
  const int lo   = lane & 15;
  const int hi   = lane >> 4;   // half-wave
  const int bw   = blockIdx.x;  // window index

  // ---- Stage A: x window -> LDS bf16, rows [49,64) zero-padded ----
  const float* xw = x + (size_t)bw * N_TOK * CDIM;
  for (int i4 = tid; i4 < NPAD * CDIM / 4; i4 += 128) {
    int row = i4 >> 5;  // (i4*4)/128
    v4f v = {0.f, 0.f, 0.f, 0.f};
    if (row < N_TOK) v = *(const v4f*)(xw + i4 * 4);
    v4bf o;
    o.x = (bf16)v.x; o.y = (bf16)v.y; o.z = (bf16)v.z; o.w = (bf16)v.w;
    *(v4bf*)(&sX[i4 * 4]) = o;
  }
  __syncthreads();

  // ---- Stage B: QKV = X(64x128) @ WqkvT(128x384); write Q/K/VT head-major ----
  {
    const int mt = wv;                       // each wave owns one 16-row block
    v16bf aX[4];
#pragma unroll
    for (int ks = 0; ks < 4; ++ks)
      aX[ks] = load_a(sX, CDIM, mt * 16, ks * 32, lo, hi);

    for (int nt = 0; nt < 24; ++nt) {        // scalar loop -> scalar seg/h/d branches
      v8f acc = v8f_zero();
#pragma unroll
      for (int ks = 0; ks < 4; ++ks)
        acc = wmma_bf16(aX[ks], load_b(wqkvT, CDIM, nt * 16, ks * 32, lo, hi), acc);

      const int   seg = nt >> 3;             // 0=q 1=k 2=v (uniform)
      const int   hh  = (nt >> 1) & 3;       // head (uniform)
      const int   d   = ((nt & 1) << 4) + lo;
      const float bv  = qkv_b[nt * 16 + lo];
      const int   m0  = mt * 16 + hi * 8;

      if (seg == 0) {
        bf16* dst = &sQ[hh][m0 * HDIM + d];
#pragma unroll
        for (int r = 0; r < 8; ++r)
          dst[r * HDIM] = (bf16)((acc[r] + bv) * QK_SCALE);
      } else if (seg == 1) {
        bf16* dst = &sK[hh][m0 * HDIM + d];
#pragma unroll
        for (int r = 0; r < 8; ++r)
          dst[r * HDIM] = (bf16)(acc[r] + bv);
      } else {
        bf16* dst = &sVT[hh][d * NPAD + m0]; // consecutive r -> consecutive m
#pragma unroll
        for (int r = 0; r < 8; r += 2) {
          v2bf pr;
          pr.x = (bf16)(acc[r] + bv);
          pr.y = (bf16)(acc[r + 1] + bv);
          *(v2bf*)(dst + r) = pr;
        }
      }
    }
  }
  __syncthreads();

  // ---- Stage C/D: per-head attention; wave wv owns head wv ----
  {
    const int h = wv;
    const float* combp = cmb + ((size_t)(bw & (NW - 1)) * HEADS + h) * NN;
    bf16* sP = sX + wv * 16 * NPAD;          // per-wave 16x64 P scratch (X dead)

    for (int mt = 0; mt < 4; ++mt) {
      // S row-block = Q[mt] (16x32) x K^T : one WMMA per 16x16 tile (K=32=d)
      v16bf aQ = load_a(sQ[h], HDIM, mt * 16, 0, lo, hi);
      v8f s[4];
#pragma unroll
      for (int nt = 0; nt < 4; ++nt)
        s[nt] = wmma_bf16(aQ, load_b(sK[h], HDIM, nt * 16, 0, lo, hi), v8f_zero());

      // combined bias/mask add + row softmax (rows r+8*hi; cols over 16 lanes x 4 tiles)
#pragma unroll
      for (int r = 0; r < 8; ++r) {
        int m = mt * 16 + r + hi * 8;
        bool mrow = (m < N_TOK);
        float vv[4];
#pragma unroll
        for (int nt = 0; nt < 4; ++nt) {
          int c = nt * 16 + lo;
          float v = s[nt][r];
          if (c >= N_TOK)   v = -1e30f;
          else if (mrow)    v += combp[m * N_TOK + c];
          vv[nt] = v;
        }
        float mx = fmaxf(fmaxf(vv[0], vv[1]), fmaxf(vv[2], vv[3]));
#pragma unroll
        for (int off = 1; off < 16; off <<= 1)
          mx = fmaxf(mx, __shfl_xor(mx, off, 32));
        float sum = 0.f;
#pragma unroll
        for (int nt = 0; nt < 4; ++nt) {
          vv[nt] = __expf(vv[nt] - mx);
          sum += vv[nt];
        }
#pragma unroll
        for (int off = 1; off < 16; off <<= 1)
          sum += __shfl_xor(sum, off, 32);
        float inv = 1.f / sum;
        int rl = r + hi * 8;
#pragma unroll
        for (int nt = 0; nt < 4; ++nt)
          sP[rl * NPAD + nt * 16 + lo] = (bf16)(vv[nt] * inv);
      }

      // O row-block = P(16x64) @ V(64x32); overwrite consumed Q rows (head-major O)
      v8f o0 = v8f_zero(), o1 = v8f_zero();
#pragma unroll
      for (int ks = 0; ks < 2; ++ks) {
        v16bf aP = load_a(sP, NPAD, 0, ks * 32, lo, hi);
        o0 = wmma_bf16(aP, load_b(sVT[h], NPAD, 0,  ks * 32, lo, hi), o0);
        o1 = wmma_bf16(aP, load_b(sVT[h], NPAD, 16, ks * 32, lo, hi), o1);
      }
      bf16* dst = &sQ[h][(mt * 16 + hi * 8) * HDIM];
#pragma unroll
      for (int r = 0; r < 8; ++r) {
        dst[r * HDIM + lo]      = (bf16)o0[r];
        dst[r * HDIM + 16 + lo] = (bf16)o1[r];
      }
    }
  }
  __syncthreads();

  // ---- Stage E: out = O(64x128 head-major) @ WprojT + proj_b ----
  {
    const int mt = wv;
    v16bf aO[4];
#pragma unroll
    for (int ks = 0; ks < 4; ++ks)           // k-step 32 == one head slice of O
      aO[ks] = load_a(sQ[ks], HDIM, mt * 16, 0, lo, hi);

    float* outw = out + (size_t)bw * N_TOK * CDIM;
    for (int nt = 0; nt < 8; ++nt) {
      v8f acc = v8f_zero();
#pragma unroll
      for (int ks = 0; ks < 4; ++ks)
        acc = wmma_bf16(aO[ks], load_b(wprojT, CDIM, nt * 16, ks * 32, lo, hi), acc);
      int col = nt * 16 + lo;
      float pb = proj_b[col];
#pragma unroll
      for (int r = 0; r < 8; ++r) {
        int m = mt * 16 + r + hi * 8;
        if (m < N_TOK)
          outw[m * CDIM + col] = acc[r] + pb;
      }
    }
  }
}

extern "C" void kernel_launch(void* const* d_in, const int* in_sizes, int n_in,
                              void* d_out, int out_size, void* d_ws, size_t ws_size,
                              hipStream_t stream) {
  const float* x          = (const float*)d_in[0];
  const float* mask       = (const float*)d_in[1];
  const float* qkv_w      = (const float*)d_in[2];
  const float* qkv_b      = (const float*)d_in[3];
  const float* proj_w     = (const float*)d_in[4];
  const float* proj_b     = (const float*)d_in[5];
  const float* bias_table = (const float*)d_in[6];
  const int*   rel_index  = (const int*)d_in[7];

  char* ws      = (char*)d_ws;
  bf16* wqkvT   = (bf16*)(ws);             // 98304 B
  bf16* wprojT  = (bf16*)(ws + 98304);     // 32768 B
  float* cmb    = (float*)(ws + 131072);   // 2458624 B

  int prep_n = 384 * 128 + 128 * 128 + NW * HEADS * NN;
  prep_kernel<<<(prep_n + 255) / 256, 256, 0, stream>>>(
      qkv_w, proj_w, bias_table, rel_index, mask, wqkvT, wprojT, cmb);

  win_attn_kernel<<<BWIN, 128, 0, stream>>>(
      x, qkv_b, proj_b, wqkvT, wprojT, cmb, (float*)d_out);
}
